// mlp_84748294685180
// MI455X (gfx1250) — matrix operand order repaired and checked by reading, not compiled, not run on hardware
//
#include <hip/hip_runtime.h>

typedef _Float16 v16h __attribute__((ext_vector_type(16)));
typedef _Float16 h8   __attribute__((ext_vector_type(8)));
typedef float    v8f  __attribute__((ext_vector_type(8)));

#define LW 128
#define IN_DIM 16
#define NB_X 4096
#define NB_THETA 128
#define PARAM_COUNT 51841
#define OFF_B0 2048
#define OFF_W1 2176
#define LAYER_STRIDE 16512
#define OFF_WL 51712
#define OFF_BL 51840

// Prefer the gfx1250 hardware tanh (v_tanh_f32); fall back to exp2+rcp.
__device__ __forceinline__ float fast_tanh(float v) {
#if __has_builtin(__builtin_amdgcn_tanhf)
  return __builtin_amdgcn_tanhf(v);
#else
  float e = __builtin_amdgcn_exp2f(v * 2.8853900817779268f); // 2*log2(e)
  return 1.0f - 2.0f * __builtin_amdgcn_rcpf(e + 1.0f);
#endif
}

// 16-byte copy with only 4-byte alignment assumption (theta rows are odd-length).
__device__ __forceinline__ void ld4(float* d, const float* p) {
  __builtin_memcpy(d, p, 16);
}

__global__ __launch_bounds__(256) void mlp_field_kernel(
    const float* __restrict__ x, const float* __restrict__ theta,
    float* __restrict__ out)
{
  // double-buffered activations, x-major: h[x 0..127][neuron 0..127], f16
  __shared__ __align__(16) _Float16 hbufA[LW * LW];
  __shared__ __align__(16) _Float16 hbufB[LW * LW];
  __shared__ float wl_sh[LW];

  const int t  = blockIdx.y;           // theta index
  const int n0 = blockIdx.x * LW;      // x-tile base
  const float* th = theta + (size_t)t * PARAM_COUNT;

  const int tid  = threadIdx.x;
  const int w    = tid >> 5;           // wave id == M-tile (output-neuron tile)
  const int lane = tid & 31;
  const int kh   = lane >> 4;          // lane half (WMMA K/M split)
  const int lm   = lane & 15;

  // ---------------- layer 0: h = tanh(W0(128x16) @ x^T(16x128) + b0), K padded 16->32
  {
    float bb[8];
    ld4(bb,     th + OFF_B0 + 16 * w + 8 * kh);
    ld4(bb + 4, th + OFF_B0 + 16 * w + 8 * kh + 4);

    // A-frag: row m = 16*w+lm; elements 0..7 -> K = 8*kh+e (valid, K<16); 8..15 -> K>=16 (zero pad)
    float wr[8];
    ld4(wr,     th + (16 * w + lm) * IN_DIM + 8 * kh);
    ld4(wr + 4, th + (16 * w + lm) * IN_DIM + 8 * kh + 4);
    v16h afrag;
    #pragma unroll
    for (int i = 0; i < 8; ++i)  afrag[i] = (_Float16)wr[i];
    #pragma unroll
    for (int i = 8; i < 16; ++i) afrag[i] = (_Float16)0.0f;

    // warm L2/WGP$ for layer-1 weights while we do layer 0
    __builtin_prefetch(th + OFF_W1 + (16 * w + lm) * LW, 0, 1);

    #pragma unroll
    for (int nt = 0; nt < 8; ++nt) {
      v8f acc;
      #pragma unroll
      for (int r = 0; r < 8; ++r) acc[r] = bb[r];

      // B-frag: elements 0..7 -> K = 8*kh + e (x features); 8..15 -> K>=16 (zero pad)
      v16h bfrag;
      {
        float xr[8];
        const float* xp = x + (size_t)(n0 + 16 * nt + lm) * IN_DIM + 8 * kh;
        ld4(xr, xp); ld4(xr + 4, xp + 4);
        #pragma unroll
        for (int e = 0; e < 8; ++e) bfrag[e] = (_Float16)xr[e];
        #pragma unroll
        for (int e = 8; e < 16; ++e) bfrag[e] = (_Float16)0.0f;
      }

      acc = __builtin_amdgcn_wmma_f32_16x16x32_f16(false, afrag, false, bfrag,
                                                   (short)0, acc, false, false);

      // D element r -> neuron M = 16*w + 8*kh + r, x N = 16*nt + lm; store x-major
      h8 st;
      #pragma unroll
      for (int r = 0; r < 8; ++r) st[r] = (_Float16)fast_tanh(acc[r]);
      *(h8*)(&hbufA[(16 * nt + lm) * LW + 16 * w + 8 * kh]) = st;
    }
  }
  __syncthreads();

  // ---------------- layers 1..3: h = tanh(W(128x128) @ h + b)
  _Float16* hcur  = hbufA;
  _Float16* hnext = hbufB;
  #pragma unroll 1
  for (int layer = 1; layer < 4; ++layer) {
    const float* wp = th + OFF_W1 + (layer - 1) * LAYER_STRIDE;
    const float* bp = wp + LW * LW;

    float bb[8];
    ld4(bb,     bp + 16 * w + 8 * kh);
    ld4(bb + 4, bp + 16 * w + 8 * kh + 4);

    // Cache 4 A-frags (K = 0..127 in steps of 32) for this wave's 16 output rows
    v16h afr[4];
    const float* wrow = wp + (16 * w + lm) * LW;
    #pragma unroll
    for (int k = 0; k < 4; ++k) {
      float q[16];
      ld4(q,      wrow + 32 * k + 8 * kh);        // e 0..7  : K = 32k + 8*kh + e
      ld4(q + 4,  wrow + 32 * k + 8 * kh + 4);
      ld4(q + 8,  wrow + 32 * k + 16 + 8 * kh);   // e 8..15 : K = 32k + 16 + 8*kh + (e-8)
      ld4(q + 12, wrow + 32 * k + 16 + 8 * kh + 4);
      #pragma unroll
      for (int e = 0; e < 16; ++e) afr[k][e] = (_Float16)q[e];
    }

    // warm next layer's weight row
    if (layer < 3)
      __builtin_prefetch(wp + LAYER_STRIDE + (16 * w + lm) * LW, 0, 1);

    #pragma unroll
    for (int nt = 0; nt < 8; ++nt) {
      v8f acc;
      #pragma unroll
      for (int r = 0; r < 8; ++r) acc[r] = bb[r];

      #pragma unroll
      for (int k = 0; k < 4; ++k) {
        // B element e -> K = 32k + 8*kh + e (e<8), 32k + 16 + 8*kh + (e-8) (e>=8), N = 16*nt + lm
        const _Float16* hp = hcur + (16 * nt + lm) * LW + 32 * k + 8 * kh;
        union { v16h v; h8 h2[2]; } bu;
        bu.h2[0] = *(const h8*)(hp);
        bu.h2[1] = *(const h8*)(hp + 16);
        acc = __builtin_amdgcn_wmma_f32_16x16x32_f16(false, afr[k], false, bu.v,
                                                     (short)0, acc, false, false);
      }

      h8 st;
      #pragma unroll
      for (int r = 0; r < 8; ++r) st[r] = (_Float16)fast_tanh(acc[r]);
      *(h8*)(&hnext[(16 * nt + lm) * LW + 16 * w + 8 * kh]) = st;
    }
    __syncthreads();
    _Float16* tmp = hcur; hcur = hnext; hnext = tmp;
  }
  // after 3 swaps the final activations are in hcur

  // ---------------- head: out[x] = Wl . h[x] + bl
  if (tid < LW) wl_sh[tid] = th[OFF_WL + tid];
  __syncthreads();
  if (tid < LW) {
    const _Float16* hr = hcur + tid * LW;
    float s = 0.0f;
    #pragma unroll 16
    for (int i = 0; i < LW; ++i) s += wl_sh[i] * (float)hr[i];
    out[(size_t)t * NB_X + n0 + tid] = s + th[OFF_BL];
  }
}

extern "C" void kernel_launch(void* const* d_in, const int* in_sizes, int n_in,
                              void* d_out, int out_size, void* d_ws, size_t ws_size,
                              hipStream_t stream) {
  const float* x     = (const float*)d_in[0];  // (4096, 16) f32
  const float* theta = (const float*)d_in[1];  // (128, 51841) f32
  float* out = (float*)d_out;                  // (128, 4096, 1) f32

  dim3 grid(NB_X / LW, NB_THETA);              // 32 x-tiles x 128 thetas
  mlp_field_kernel<<<grid, 256, 0, stream>>>(x, theta, out);
}
